// MultiHeadAttentionRope_82008105550575
// MI455X (gfx1250) — matrix operand order repaired and checked by reading, not compiled, not run on hardware
//
#include <hip/hip_runtime.h>
#include <hip/hip_bf16.h>
#include <cstdint>

// ---------- types ----------
typedef __hip_bfloat16 hbf;                                    // storage type
typedef __attribute__((ext_vector_type(16))) __bf16 v16bf;     // WMMA A/B fragment
typedef __attribute__((ext_vector_type(8)))  __bf16 v8bf;      // half-fragment chunk
typedef __attribute__((ext_vector_type(8)))  float  v8f;       // WMMA C/D fragment
typedef __attribute__((ext_vector_type(16))) unsigned short v16us;
typedef __attribute__((ext_vector_type(4)))  unsigned int u32x4;
typedef __attribute__((ext_vector_type(8)))  int i32x8;
typedef __attribute__((ext_vector_type(4)))  int i32x4;

#define WMMA_BF16(a, b, c) \
  __builtin_amdgcn_wmma_f32_16x16x32_bf16(false, (a), false, (b), (short)0, (c), false, false)

#define B_  2
#define S_  2048
#define D_  1024
#define H_  16
#define DK_ 64
#define M_  (B_ * S_)   // 4096 rows

// ---------- fragment loaders (contiguous-k layouts, A and B alike, wave32) ----------
// A (16xK row-major, 16-bit): lane m=(lane&15); lanes<16 hold K 0..15,
// lanes>=16 hold K 16..31 of the current 32-wide K step.
__device__ __forceinline__ v16bf load_a_frag(const hbf* base, int ldk, int lane) {
  int m   = lane & 15;
  int klo = (lane < 16) ? 0 : 16;
  const __bf16* p = (const __bf16*)(base + (size_t)m * ldk + klo);
  v8bf lo = *(const v8bf*)(p);
  v8bf hi = *(const v8bf*)(p + 8);
  v16bf r;
#pragma unroll
  for (int i = 0; i < 8; ++i) { r[i] = lo[i]; r[8 + i] = hi[i]; }
  return r;
}

// B (32x16, 16-bit): lane holds column n=(lane&15), K contiguous:
// lanes<16 K=0..15, lanes>=16 K=16..31. Column n contiguous in memory,
// stride ldk between columns.
__device__ __forceinline__ v16bf load_b_frag(const hbf* colbase, int ldk, int lane) {
  int n  = lane & 15;
  int ks = (lane < 16) ? 0 : 16;
  return *(const v16bf*)(colbase + (size_t)n * ldk + ks);
}

// ---------- Tensor Data Mover: 2-D strided global tile -> contiguous LDS ----------
// D# per CDNA5 ISA section 8: group0 {count,lds_addr,global_addr,type=2},
// group1 {data_size=2B, tensor dims == tile dims, dim0 stride}. Groups 2/3 zero
// (<=2-D tensor). One TENSORcnt per issue.
__device__ __forceinline__ void tdm_load_2d(unsigned lds_off, const void* gptr,
                                            unsigned d0, unsigned d1,
                                            unsigned stride_elems) {
  unsigned long long ga = (unsigned long long)(uintptr_t)gptr;
  u32x4 g0;
  g0[0] = 1u;                                         // count=1 (user descriptor)
  g0[1] = lds_off;                                    // lds_addr (bytes)
  g0[2] = (unsigned)(ga & 0xFFFFFFFFull);             // global_addr[31:0]
  g0[3] = (unsigned)((ga >> 32) & 0x1FFFFFFull) | (2u << 30);  // addr[56:32] | type=2
  i32x8 g1;
  g1[0] = 1 << 16;                                    // wg_mask=0, data_size=1 (2B)
  g1[1] = (int)((d0 & 0xFFFFu) << 16);                // tensor_dim0[15:0]
  g1[2] = (int)(((d0 >> 16) & 0xFFFFu) | ((d1 & 0xFFFFu) << 16));  // dim0_hi | dim1_lo
  g1[3] = (int)(((d1 >> 16) & 0xFFFFu) | ((d0 & 0xFFFFu) << 16));  // dim1_hi | tile_dim0
  g1[4] = (int)(d1 & 0xFFFFu);                        // tile_dim1 | tile_dim2=0
  g1[5] = (int)stride_elems;                          // tensor_dim0_stride[31:0]
  g1[6] = 0;
  g1[7] = 0;
  i32x4 z = {0, 0, 0, 0};
#if defined(__clang_major__) && (__clang_major__ >= 23)
  i32x8 z8 = {0, 0, 0, 0, 0, 0, 0, 0};
  __builtin_amdgcn_tensor_load_to_lds(g0, g1, z, z, z8, 0);
#else
  __builtin_amdgcn_tensor_load_to_lds(g0, g1, z, z, 0);
#endif
}

// ---------- elementwise: f32 -> bf16 ----------
__global__ void cvt_f32_bf16(const float* __restrict__ in, hbf* __restrict__ out, int n) {
  int i = blockIdx.x * blockDim.x + threadIdx.x;
  if (i < n) out[i] = __float2bfloat16(in[i]);
}

// ---------- GEMM: C[m,n] = sum_k A[m,k] * W[n,k]  (A: MxK, W: NxK, bf16) ----------
// wave -> 32(M) x 64(N) tile: 2 A fragments x 4 B fragments = 8 WMMA per K-step,
// 12 b128 loads per 8 WMMAs. Working set is L2-resident; stream from global.
__global__ __launch_bounds__(128) void gemm_xwT(
    const hbf* __restrict__ A, const hbf* __restrict__ W,
    hbf* __restrict__ Cb, float* __restrict__ Cf, int M, int N, int K) {
  int lane = threadIdx.x;                 // 0..31
  int wave = threadIdx.y;                 // 0..3
  int m0 = (blockIdx.y * 4 + wave) * 32;
  int n0 = blockIdx.x * 64;
  if (m0 >= M) return;                    // wave-uniform

  v8f acc[2][4] = {};
  for (int k0 = 0; k0 < K; k0 += 32) {
    v16bf a0 = load_a_frag(A + (size_t)m0 * K + k0, K, lane);
    v16bf a1 = load_a_frag(A + (size_t)(m0 + 16) * K + k0, K, lane);
#pragma unroll
    for (int j = 0; j < 4; ++j) {
      v16bf b = load_b_frag(W + (size_t)(n0 + j * 16) * K + k0, K, lane);
      acc[0][j] = WMMA_BF16(a0, b, acc[0][j]);
      acc[1][j] = WMMA_BF16(a1, b, acc[1][j]);
    }
  }
  int n = lane & 15;
#pragma unroll
  for (int h2 = 0; h2 < 2; ++h2) {
    int mb = m0 + h2 * 16 + ((lane < 16) ? 0 : 8);
#pragma unroll
    for (int j = 0; j < 4; ++j) {
      int col = n0 + j * 16 + n;
#pragma unroll
      for (int r = 0; r < 8; ++r) {
        if (Cf) Cf[(size_t)(mb + r) * N + col] = acc[h2][j][r];
        else    Cb[(size_t)(mb + r) * N + col] = __float2bfloat16(acc[h2][j][r]);
      }
    }
  }
}

// ---------- RoPE in-place on bf16 Q and K ----------
__global__ void rope_kernel(hbf* __restrict__ q, hbf* __restrict__ k,
                            const int* __restrict__ pos) {
  int idx = blockIdx.x * blockDim.x + threadIdx.x;  // pair index
  if (idx >= M_ * (D_ / 2)) return;
  int row = idx / (D_ / 2);
  int dp  = idx % (D_ / 2);
  int i   = dp & (DK_ / 2 - 1);                      // pair index within head
  float p = (float)pos[row % S_];
  float ang = p * __powf(10000.0f, -((float)(2 * i) / (float)DK_));
  float sn, cs;
  __sincosf(ang, &sn, &cs);
  size_t base = (size_t)row * D_ + 2 * dp;
  float q0 = __bfloat162float(q[base]), q1 = __bfloat162float(q[base + 1]);
  q[base]     = __float2bfloat16(q0 * cs - q1 * sn);
  q[base + 1] = __float2bfloat16(q0 * sn + q1 * cs);
  float k0 = __bfloat162float(k[base]), k1 = __bfloat162float(k[base + 1]);
  k[base]     = __float2bfloat16(k0 * cs - k1 * sn);
  k[base + 1] = __float2bfloat16(k0 * sn + k1 * cs);
}

// ---------- flash attention: block = (b, h, 64 q rows); 4 waves x 16 rows ----------
// TDM double-buffers K/V tiles into LDS; next chunk's DMA overlaps current
// chunk's WMMA + online softmax.
__global__ __launch_bounds__(128) void flash_attn(
    const hbf* __restrict__ Q, const hbf* __restrict__ Kg, const hbf* __restrict__ Vg,
    hbf* __restrict__ O) {
  int bid = blockIdx.x;
  int qt  = bid % (S_ / 64);
  int h   = (bid / (S_ / 64)) % H_;
  int b   = bid / ((S_ / 64) * H_);
  int tid  = threadIdx.x;
  int lane = tid & 31;
  int wave = tid >> 5;
  int qbase = qt * 64 + wave * 16;

  __shared__ alignas(32) hbf Kt[2][32][DK_];    // [buf][kv][dk]        8 KB
  __shared__ alignas(32) hbf Vr[2][32][DK_];    // [buf][kv][dk]        8 KB
  __shared__ alignas(32) hbf Vt[2][DK_][32];    // [buf][dk][kv] (V^T)  8 KB
  __shared__ alignas(32) hbf Pt[4][16][32];     // per-wave P           4 KB

  const size_t bS = (size_t)b * S_;
  const hbf* qptr = Q + (bS + qbase) * D_ + h * DK_;
  v16bf qa = load_a_frag(qptr,      D_, lane);  // dk 0..31
  v16bf qb = load_a_frag(qptr + 32, D_, lane);  // dk 32..63

  v8f acc[4] = {};
  float mrow[8], lrow[8];
#pragma unroll
  for (int r = 0; r < 8; ++r) { mrow[r] = -1e30f; lrow[r] = 0.0f; }
  int mlocal = (lane < 16) ? 0 : 8;
  int n = lane & 15;
  int nchunks = 2 * qt + 2;                     // causal: kv0 <= qt*64+63

  if (wave == 0) {                              // prefetch chunk 0 via TDM
    tdm_load_2d((unsigned)(uintptr_t)&Kt[0][0][0], Kg + bS * D_ + h * DK_, DK_, 32, D_);
    tdm_load_2d((unsigned)(uintptr_t)&Vr[0][0][0], Vg + bS * D_ + h * DK_, DK_, 32, D_);
  }

  for (int i = 0; i < nchunks; ++i) {
    int kv0 = i * 32;
    int buf = i & 1;
    if (wave == 0) __builtin_amdgcn_s_wait_tensorcnt((short)0);  // chunk i landed
    __syncthreads();                            // visible; prior buf readers done

    if (wave == 0 && (i + 1) < nchunks) {       // overlap next chunk's DMA
      const hbf* kn = Kg + (bS + kv0 + 32) * D_ + h * DK_;
      const hbf* vn = Vg + (bS + kv0 + 32) * D_ + h * DK_;
      tdm_load_2d((unsigned)(uintptr_t)&Kt[buf ^ 1][0][0], kn, DK_, 32, D_);
      tdm_load_2d((unsigned)(uintptr_t)&Vr[buf ^ 1][0][0], vn, DK_, 32, D_);
    }

    {   // LDS transpose Vr[buf] -> Vt[buf]: 1x ds_load_b128 pair + 16 b16 stores
      int kv = tid >> 2, seg = (tid & 3) * 16;
      v16us row = *(const v16us*)&Vr[buf][kv][seg];
#pragma unroll
      for (int t2 = 0; t2 < 16; ++t2)
        *(unsigned short*)&Vt[buf][seg + t2][kv] = row[t2];
    }
    __syncthreads();

    // scores: two 16(q) x 16(kv) tiles, K-dim = dk = 64 -> 2 WMMA each
    v8f s0 = {}, s1 = {};
    s0 = WMMA_BF16(qa, load_b_frag(&Kt[buf][0][0],   DK_, lane), s0);
    s0 = WMMA_BF16(qb, load_b_frag(&Kt[buf][0][32],  DK_, lane), s0);
    s1 = WMMA_BF16(qa, load_b_frag(&Kt[buf][16][0],  DK_, lane), s1);
    s1 = WMMA_BF16(qb, load_b_frag(&Kt[buf][16][32], DK_, lane), s1);

    const float scale = 0.125f;                  // 1/sqrt(64)
#pragma unroll
    for (int r = 0; r < 8; ++r) {
      int qidx = qbase + mlocal + r;
      float e0 = (kv0 + n      <= qidx) ? s0[r] * scale : -1e30f;
      float e1 = (kv0 + 16 + n <= qidx) ? s1[r] * scale : -1e30f;
      float cm = fmaxf(e0, e1);                  // row reduce over 16-lane half
#pragma unroll
      for (int mk = 1; mk < 16; mk <<= 1) cm = fmaxf(cm, __shfl_xor(cm, mk, 32));
      float nm   = fmaxf(mrow[r], cm);
      float corr = __expf(mrow[r] - nm);
      float p0 = __expf(e0 - nm), p1 = __expf(e1 - nm);
      float ps = p0 + p1;
#pragma unroll
      for (int mk = 1; mk < 16; mk <<= 1) ps += __shfl_xor(ps, mk, 32);
      lrow[r] = lrow[r] * corr + ps;
      mrow[r] = nm;
#pragma unroll
      for (int j = 0; j < 4; ++j) acc[j][r] *= corr;
      Pt[wave][mlocal + r][n]      = __float2bfloat16(p0);   // P in A row-major
      Pt[wave][mlocal + r][16 + n] = __float2bfloat16(p1);
    }
    __syncthreads();                             // P + all Kt[buf] reads done

    // O += P(16x32) @ V(32x64): one WMMA per 16-wide dk group
    v16bf pa = load_a_frag(&Pt[wave][0][0], 32, lane);
#pragma unroll
    for (int j = 0; j < 4; ++j)
      acc[j] = WMMA_BF16(pa, load_b_frag(&Vt[buf][j * 16][0], 32, lane), acc[j]);
  }

  // normalize and store bf16
  hbf* optr = O + (bS + qbase + mlocal) * D_ + h * DK_;
#pragma unroll
  for (int j = 0; j < 4; ++j) {
    int col = j * 16 + n;
#pragma unroll
    for (int r = 0; r < 8; ++r)
      optr[(size_t)r * D_ + col] = __float2bfloat16(acc[j][r] / lrow[r]);
  }
}

// ---------- host orchestration ----------
extern "C" void kernel_launch(void* const* d_in, const int* in_sizes, int n_in,
                              void* d_out, int out_size, void* d_ws, size_t ws_size,
                              hipStream_t stream) {
  const float* x   = (const float*)d_in[0];
  const int*   pos = (const int*)d_in[1];
  const float* wq  = (const float*)d_in[2];
  const float* wk  = (const float*)d_in[3];
  const float* wv  = (const float*)d_in[4];
  const float* wo  = (const float*)d_in[5];

  char* ws = (char*)d_ws;
  hbf* xb  = (hbf*)ws; ws += (size_t)M_ * D_ * 2;     // 8 MB
  hbf* wqb = (hbf*)ws; ws += (size_t)D_ * D_ * 2;     // 2 MB
  hbf* wkb = (hbf*)ws; ws += (size_t)D_ * D_ * 2;
  hbf* wvb = (hbf*)ws; ws += (size_t)D_ * D_ * 2;
  hbf* wob = (hbf*)ws; ws += (size_t)D_ * D_ * 2;
  hbf* Qb  = (hbf*)ws; ws += (size_t)M_ * D_ * 2;     // 8 MB
  hbf* Kb  = (hbf*)ws; ws += (size_t)M_ * D_ * 2;
  hbf* Vb  = (hbf*)ws; ws += (size_t)M_ * D_ * 2;
  hbf* Ob  = (hbf*)ws; ws += (size_t)M_ * D_ * 2;

  int nx = M_ * D_, nw = D_ * D_;
  cvt_f32_bf16<<<(nx + 255) / 256, 256, 0, stream>>>(x,  xb,  nx);
  cvt_f32_bf16<<<(nw + 255) / 256, 256, 0, stream>>>(wq, wqb, nw);
  cvt_f32_bf16<<<(nw + 255) / 256, 256, 0, stream>>>(wk, wkb, nw);
  cvt_f32_bf16<<<(nw + 255) / 256, 256, 0, stream>>>(wv, wvb, nw);
  cvt_f32_bf16<<<(nw + 255) / 256, 256, 0, stream>>>(wo, wob, nw);

  dim3 gblk(32, 4);
  dim3 ggrd(D_ / 64, M_ / 128);
  gemm_xwT<<<ggrd, gblk, 0, stream>>>(xb, wqb, Qb, nullptr, M_, D_, D_);
  gemm_xwT<<<ggrd, gblk, 0, stream>>>(xb, wkb, Kb, nullptr, M_, D_, D_);
  gemm_xwT<<<ggrd, gblk, 0, stream>>>(xb, wvb, Vb, nullptr, M_, D_, D_);

  int npair = M_ * (D_ / 2);
  rope_kernel<<<(npair + 255) / 256, 256, 0, stream>>>(Qb, Kb, pos);

  flash_attn<<<B_ * H_ * (S_ / 64), 128, 0, stream>>>(Qb, Kb, Vb, Ob);

  gemm_xwT<<<ggrd, gblk, 0, stream>>>(Ob, wob, nullptr, (float*)d_out, M_, D_, D_);
}
